// Conv1x1GnGeluSequential_2224793059392
// MI455X (gfx1250) — matrix operand order repaired and checked by reading, not compiled, not run on hardware
//
#include <hip/hip_runtime.h>
#include <math.h>

typedef __attribute__((ext_vector_type(16))) __bf16 v16bf;
typedef __attribute__((ext_vector_type(8)))  __bf16 v8bf;
typedef __attribute__((ext_vector_type(8)))  float  v8f;
typedef int v4i __attribute__((vector_size(16)));   // b128 payload type for async LDS builtins

#define C_DIM 256
#define BATCHES 8
#define GROUPS 32
#define CPG 8
#define EPSV 1e-5f
#define ROW_PAD 40   // 32 bf16 + 8 pad per staged W row (80B stride, kills bank conflicts)

// gfx1250 async global->LDS path (ASYNCcnt-tracked), guarded so the file
// compiles on toolchains where the builtin is absent.
#if defined(__has_builtin)
#  if __has_builtin(__builtin_amdgcn_global_load_async_to_lds_b128) && \
      __has_builtin(__builtin_amdgcn_s_wait_asynccnt)
#    define USE_ASYNC_LDS 1
#  endif
#endif
#ifndef USE_ASYNC_LDS
#  define USE_ASYNC_LDS 0
#endif

#define ASYNC_GPTR(p) ((__attribute__((address_space(1))) v4i*)(p))
#define ASYNC_LPTR(p) ((__attribute__((address_space(3))) v4i*)(p))

// ---------------- init: zero stat accumulators ----------------
__global__ void init_kernel(float* gS1, float* gS2, float* counts) {
    int i = threadIdx.x;
    for (int k = i; k < BATCHES * C_DIM; k += 256) { gS1[k] = 0.f; gS2[k] = 0.f; }
    if (i < BATCHES) counts[i] = 0.f;
}

// ---------------- split W into bf16 hi/lo ----------------
__global__ void wsplit_kernel(const float* __restrict__ w, __bf16* __restrict__ whi,
                              __bf16* __restrict__ wlo) {
    int i = blockIdx.x * 256 + threadIdx.x;           // 65536 elements
    float f = w[i];
    __bf16 h = (__bf16)f;
    whi[i] = h;
    wlo[i] = (__bf16)(f - (float)h);
}

// ---------------- rows-per-batch histogram ----------------
__global__ void count_kernel(const int* __restrict__ bid, float* counts, int n) {
    __shared__ int hist[BATCHES];
    if (threadIdx.x < BATCHES) hist[threadIdx.x] = 0;
    __syncthreads();
    int i = blockIdx.x * blockDim.x + threadIdx.x;
    if (i < n) atomicAdd(&hist[bid[i]], 1);
    __syncthreads();
    if (threadIdx.x < BATCHES && hist[threadIdx.x])
        atomicAdd(&counts[threadIdx.x], (float)hist[threadIdx.x]);
}

// load 16 bf16 (two 16B-aligned halves, at p and p+16) from LDS
__device__ __forceinline__ v16bf load_v16bf(const __bf16* p) {
    v8bf a = *(const v8bf*)p;
    v8bf b = *(const v8bf*)(p + 16);
    v16bf r;
#pragma unroll
    for (int i = 0; i < 8; ++i) { r[i] = a[i]; r[8 + i] = b[i]; }
    return r;
}

// stage one 256x32 bf16 k-slice of W_hi and W_lo into LDS (thread t copies row o=t)
__device__ __forceinline__ void stage_slice(const __bf16* __restrict__ whi,
                                            const __bf16* __restrict__ wlo,
                                            __bf16* dWhi, __bf16* dWlo,
                                            int tid, int kb) {
    const __bf16* sh = whi + tid * C_DIM + kb * 32;
    const __bf16* sl = wlo + tid * C_DIM + kb * 32;
    __bf16* dh = dWhi + tid * ROW_PAD;
    __bf16* dl = dWlo + tid * ROW_PAD;
#if USE_ASYNC_LDS
#pragma unroll
    for (int j = 0; j < 4; ++j) {
        __builtin_amdgcn_global_load_async_to_lds_b128(
            ASYNC_GPTR(sh + j * 8), ASYNC_LPTR(dh + j * 8), 0, 0);
        __builtin_amdgcn_global_load_async_to_lds_b128(
            ASYNC_GPTR(sl + j * 8), ASYNC_LPTR(dl + j * 8), 0, 0);
    }
#else
    const uint4* s0 = (const uint4*)sh;
    const uint4* s1 = (const uint4*)sl;
    uint4* d0 = (uint4*)dh;
    uint4* d1 = (uint4*)dl;
    d0[0] = s0[0]; d0[1] = s0[1]; d0[2] = s0[2]; d0[3] = s0[3];
    d1[0] = s1[0]; d1[1] = s1[1]; d1[2] = s1[2]; d1[3] = s1[3];
#endif
}

// ---------------- GEMM (bf16 hi/lo WMMA) + per-(batch,channel) stats ----------------
// Block: 256 threads = 8 waves. Block tile: 128 rows x 256 cols. Wave tile: 16 rows x 256 cols.
// Double-buffered async staging: slice k+1 copies overlap compute of slice k.
__global__ __launch_bounds__(256)
void gemm_stats_kernel(const float* __restrict__ x, const __bf16* __restrict__ whi,
                       const __bf16* __restrict__ wlo, const int* __restrict__ bid,
                       float* __restrict__ hout, float* __restrict__ gS1,
                       float* __restrict__ gS2) {
    __shared__ __bf16 sW[2][2][C_DIM * ROW_PAD];   // [buf][hi/lo], 80 KB
    __shared__ float  sS1[BATCHES * C_DIM];        // 8 KB
    __shared__ float  sS2[BATCHES * C_DIM];        // 8 KB

    const int tid  = threadIdx.x;
    const int lane = tid & 31;
    const int wid  = tid >> 5;
    const int blockRow = blockIdx.x * 128;
    const int waveRow  = blockRow + wid * 16;

    for (int i = tid; i < BATCHES * C_DIM; i += 256) { sS1[i] = 0.f; sS2[i] = 0.f; }

    v8f acc[16] = {};

    // A-fragment addressing (16-bit A 16x32 layout, ISA 7.12.2):
    // lanes 0-15: row M=lane, K = {k..k+7, k+16..k+23}; lanes 16-31: row M=lane-16, K += 8
    const float* xrow = x + (size_t)(waveRow + (lane & 15)) * C_DIM + ((lane >> 4) << 3);
    const int nlane = lane & 15;
    const int ksel  = (lane >> 4) << 3;   // B: lanes 0-15 hold K {0..7, 16..23}, lanes 16-31 hold K {8..15, 24..31}

    stage_slice(whi, wlo, sW[0][0], sW[0][1], tid, 0);   // prologue: slice 0 -> buf 0

    for (int kb = 0; kb < 8; ++kb) {
        const int cur = kb & 1;
#if USE_ASYNC_LDS
        __builtin_amdgcn_s_wait_asynccnt(0);   // my slice-kb copies landed
#endif
        // barrier: slice kb visible to all AND all waves done reading buf cur^1 (compute kb-1)
        __syncthreads();
        if (kb < 7)
            stage_slice(whi, wlo, sW[cur ^ 1][0], sW[cur ^ 1][1], tid, kb + 1);

        // A fragment: fp32 -> bf16 hi + residual lo
        v16bf ahi, alo;
        const float* c0 = xrow + kb * 32;
#pragma unroll
        for (int i = 0; i < 8; ++i) {
            float f0 = c0[i];
            float f1 = c0[16 + i];
            __bf16 h0 = (__bf16)f0; ahi[i] = h0;     alo[i] = (__bf16)(f0 - (float)h0);
            __bf16 h1 = (__bf16)f1; ahi[8 + i] = h1; alo[8 + i] = (__bf16)(f1 - (float)h1);
        }

        const __bf16* baseH = sW[cur][0] + nlane * ROW_PAD + ksel;
        const __bf16* baseL = sW[cur][1] + nlane * ROW_PAD + ksel;
#pragma unroll
        for (int t = 0; t < 16; ++t) {
            v16bf bhi = load_v16bf(baseH + t * 16 * ROW_PAD);
            v16bf blo = load_v16bf(baseL + t * 16 * ROW_PAD);
            acc[t] = __builtin_amdgcn_wmma_f32_16x16x32_bf16(false, ahi, false, bhi,
                                                             (short)0, acc[t], false, false);
            acc[t] = __builtin_amdgcn_wmma_f32_16x16x32_bf16(false, ahi, false, blo,
                                                             (short)0, acc[t], false, false);
            acc[t] = __builtin_amdgcn_wmma_f32_16x16x32_bf16(false, alo, false, bhi,
                                                             (short)0, acc[t], false, false);
        }
    }

    // Epilogue: C/D layout (32-bit 16x16): lane n = lane&15; VGPR j -> row j + (lane>=16 ? 8 : 0)
    const int rbase = waveRow + ((lane >> 4) << 3);
    const int bF = bid[rbase];
    const int bL = bid[rbase + 7];
    const bool uni = (bF == bL);   // batch_id sorted -> all 8 rows share bid

#pragma unroll
    for (int t = 0; t < 16; ++t) {
        const int ch = t * 16 + nlane;
        float* orow = hout + (size_t)rbase * C_DIM + ch;
        if (uni) {
            float s1 = 0.f, s2 = 0.f;
#pragma unroll
            for (int j = 0; j < 8; ++j) {
                float v = acc[t][j];
                orow[(size_t)j * C_DIM] = v;
                s1 += v; s2 += v * v;
            }
            atomicAdd(&sS1[bF * C_DIM + ch], s1);
            atomicAdd(&sS2[bF * C_DIM + ch], s2);
        } else {
#pragma unroll
            for (int j = 0; j < 8; ++j) {
                float v = acc[t][j];
                orow[(size_t)j * C_DIM] = v;
                int b = bid[rbase + j];
                atomicAdd(&sS1[b * C_DIM + ch], v);
                atomicAdd(&sS2[b * C_DIM + ch], v * v);
            }
        }
    }
    __syncthreads();

    // flush only the batches this block touches (sorted -> contiguous range)
    const int b0 = bid[blockRow];
    const int b1 = bid[blockRow + 127];
    for (int b = b0; b <= b1; ++b) {
        float v1 = sS1[b * C_DIM + tid];
        float v2 = sS2[b * C_DIM + tid];
        if (v1 != 0.f || v2 != 0.f) {
            atomicAdd(&gS1[b * C_DIM + tid], v1);
            atomicAdd(&gS2[b * C_DIM + tid], v2);
        }
    }
}

// ---------------- per-(batch,group) mean / inv_std ----------------
__global__ void finalize_kernel(const float* __restrict__ gS1, const float* __restrict__ gS2,
                                const float* __restrict__ counts, float* __restrict__ mg,
                                float* __restrict__ istd) {
    int tid = threadIdx.x;            // 256 = 8 batches x 32 groups
    int b = tid >> 5, g = tid & 31;
    float s1 = 0.f, s2 = 0.f;
    for (int c = 0; c < CPG; ++c) {
        s1 += gS1[b * C_DIM + g * CPG + c];
        s2 += gS2[b * C_DIM + g * CPG + c];
    }
    float cnt = counts[b] * (float)CPG;
    float ic  = 1.0f / (cnt + EPSV);
    float m   = s1 * ic;
    float var = ic * (s2 - 2.0f * m * s1 + cnt * m * m);
    mg[tid]   = m;
    istd[tid] = rsqrtf(var + EPSV);
}

// ---------------- normalize + affine + exact GELU (in place on d_out) ----------------
__global__ __launch_bounds__(256)
void norm_gelu_kernel(float* __restrict__ h, const int* __restrict__ bid,
                      const float* __restrict__ gn_w, const float* __restrict__ gn_b,
                      const float* __restrict__ mg, const float* __restrict__ istd) {
    size_t idx = (size_t)blockIdx.x * 256 + threadIdx.x;   // N*64 threads, float4 each
    size_t row = idx >> 6;
    int c4 = (int)(idx & 63) << 2;                         // 4 channels, same group (cpg=8)
    int b = bid[row];
    int g = c4 >> 3;
    float m  = mg[b * GROUPS + g];
    float is = istd[b * GROUPS + g];
    float4 v = *(float4*)(h + row * C_DIM + c4);
    float4 w = *(const float4*)(gn_w + c4);
    float4 bb = *(const float4*)(gn_b + c4);
    float o0 = (v.x - m) * is * w.x + bb.x;
    float o1 = (v.y - m) * is * w.y + bb.y;
    float o2 = (v.z - m) * is * w.z + bb.z;
    float o3 = (v.w - m) * is * w.w + bb.w;
    const float k = 0.70710678118654752f;
    v.x = 0.5f * o0 * (1.0f + erff(o0 * k));
    v.y = 0.5f * o1 * (1.0f + erff(o1 * k));
    v.z = 0.5f * o2 * (1.0f + erff(o2 * k));
    v.w = 0.5f * o3 * (1.0f + erff(o3 * k));
    *(float4*)(h + row * C_DIM + c4) = v;
}

extern "C" void kernel_launch(void* const* d_in, const int* in_sizes, int n_in,
                              void* d_out, int out_size, void* d_ws, size_t ws_size,
                              hipStream_t stream) {
    const float* x      = (const float*)d_in[0];
    const float* conv_w = (const float*)d_in[1];
    const float* gn_w   = (const float*)d_in[2];
    const float* gn_b   = (const float*)d_in[3];
    const int*   bid    = (const int*)d_in[4];
    float* out = (float*)d_out;
    const int N = in_sizes[0] / C_DIM;   // 262144

    // workspace carve-out (~280 KB)
    char* ws = (char*)d_ws;
    __bf16* whi = (__bf16*)ws;  ws += C_DIM * C_DIM * sizeof(__bf16);
    __bf16* wlo = (__bf16*)ws;  ws += C_DIM * C_DIM * sizeof(__bf16);
    float* gS1    = (float*)ws; ws += BATCHES * C_DIM * sizeof(float);
    float* gS2    = (float*)ws; ws += BATCHES * C_DIM * sizeof(float);
    float* counts = (float*)ws; ws += 16 * sizeof(float);
    float* mg     = (float*)ws; ws += BATCHES * GROUPS * sizeof(float);
    float* istd   = (float*)ws; ws += BATCHES * GROUPS * sizeof(float);

    init_kernel<<<1, 256, 0, stream>>>(gS1, gS2, counts);
    wsplit_kernel<<<(C_DIM * C_DIM) / 256, 256, 0, stream>>>(conv_w, whi, wlo);
    count_kernel<<<(N + 255) / 256, 256, 0, stream>>>(bid, counts, N);
    gemm_stats_kernel<<<N / 128, 256, 0, stream>>>(x, whi, wlo, bid, out, gS1, gS2);
    finalize_kernel<<<1, 256, 0, stream>>>(gS1, gS2, counts, mg, istd);
    norm_gelu_kernel<<<(size_t)N * 64 / 256, 256, 0, stream>>>(out, bid, gn_w, gn_b, mg, istd);
}
